// SAGE_3504693313813
// MI455X (gfx1250) — matrix operand order repaired and checked by reading, not compiled, not run on hardware
//
#include <hip/hip_runtime.h>
#include <hip/hip_bf16.h>
#include <math.h>

typedef __attribute__((ext_vector_type(16))) __bf16       v16bf;
typedef __attribute__((ext_vector_type(8)))  float        v8f;
typedef __attribute__((ext_vector_type(4)))  unsigned int v4u;

#define TM 128
#define TN 64
#define TK 64
#define LDA 72   // padded bf16 elems per LDS row (144B, 16B-aligned rows)
#define LDB 72

__device__ __forceinline__ unsigned short f2bf(float x) {
    __bf16 b = (__bf16)x;                       // hardware cvt (RNE)
    return __builtin_bit_cast(unsigned short, b);
}

// ---------------------------------------------------------------------------
// bf16 GEMM: C[M,N] = A1[M,K]@B1t^T (+ A2[M,K]@B2t^T if A2) + bias, opt relu.
// A row-major bf16 [M,K]; Bt is the weight pre-transposed bf16 [N,K].
// K multiple of 64, N multiple of 64. Software-pipelined via register prefetch;
// dual-matrix K loop advances raw pointers (no div/mod in the hot loop).
// ---------------------------------------------------------------------------
__global__ __launch_bounds__(256) void gemm_bf16_wmma_kernel(
    const unsigned short* __restrict__ A1, const unsigned short* __restrict__ B1t,
    const unsigned short* __restrict__ A2, const unsigned short* __restrict__ B2t,
    const float* __restrict__ bias, float* __restrict__ C,
    int M, int K, int N, int relu)
{
    __shared__ unsigned short sA[TM * LDA];   // [m][k]
    __shared__ unsigned short sB[TN * LDB];   // [n][k]

    const int tid  = threadIdx.x;
    const int wave = tid >> 5;                // 0..7 -> rows wave*16..+15
    const int lane = tid & 31;
    const int g    = lane >> 4;               // lane half
    const int lm   = lane & 15;
    const long long rowBase = (long long)blockIdx.x * TM;
    const int colBase = blockIdx.y * TN;

    // staging coords: A 128x64 -> 2 thr/row x 32 elems; B 64x64 -> 4 thr/row x 16
    const int aR = tid >> 1;
    const int aH = (tid & 1) << 5;            // 0 or 32 elems
    long long aRow = rowBase + aR;
    if (aRow >= M) aRow = M - 1;              // clamp: branch-free, masked at store
    const int bR = tid >> 2;
    const int bQ = (tid & 3) << 4;            // 0,16,32,48 elems

    const int tilesPerMat = K / TK;
    const int nTiles = (A2 ? 2 : 1) * tilesPerMat;

    // per-thread tile pointers, stepped by TK per iteration
    const unsigned short* ap  = A1 + aRow * K + aH;
    const unsigned short* bp  = B1t + (long long)(colBase + bR) * K + bQ;
    const unsigned short* ap2 = A2 ? A2 + aRow * K + aH : nullptr;
    const unsigned short* bp2 = A2 ? B2t + (long long)(colBase + bR) * K + bQ : nullptr;

    v8f acc[4] = {};
    v4u pa[4], pb[2];

    // prefetch tile 0
    {
        const v4u* a4 = (const v4u*)ap;
        pa[0] = a4[0]; pa[1] = a4[1]; pa[2] = a4[2]; pa[3] = a4[3];
        const v4u* b4 = (const v4u*)bp;
        pb[0] = b4[0]; pb[1] = b4[1];
    }

    for (int t = 0; t < nTiles; ++t) {
        // commit staged registers to LDS
        {
            v4u* sa = (v4u*)&sA[aR * LDA + aH];
            sa[0] = pa[0]; sa[1] = pa[1]; sa[2] = pa[2]; sa[3] = pa[3];
            v4u* sb = (v4u*)&sB[bR * LDB + bQ];
            sb[0] = pb[0]; sb[1] = pb[1];
        }
        __syncthreads();

        if (t + 1 < nTiles) {                  // overlap next global load w/ math
            if (t + 1 == tilesPerMat) { ap = ap2; bp = bp2; }
            else                      { ap += TK; bp += TK; }
            const v4u* a4 = (const v4u*)ap;
            pa[0] = a4[0]; pa[1] = a4[1]; pa[2] = a4[2]; pa[3] = a4[3];
            const v4u* b4 = (const v4u*)bp;
            pb[0] = b4[0]; pb[1] = b4[1];
        }

        #pragma unroll
        for (int ks = 0; ks < 2; ++ks) {
            // A fragment (16x32 ISA layout): b128 at K=g*8, second at 16+g*8
            union { v16bf v; v4u q[2]; } af;
            const unsigned short* arow = &sA[(wave * 16 + lm) * LDA + ks * 32];
            af.q[0] = *(const v4u*)(arow + g * 8);
            af.q[1] = *(const v4u*)(arow + 16 + g * 8);
            // preload all B fragments, then issue WMMAs back-to-back
            v4u bq[8];
            #pragma unroll
            for (int nb = 0; nb < 4; ++nb) {
                const unsigned short* brow = &sB[(nb * 16 + lm) * LDB + ks * 32];
                bq[2 * nb]     = *(const v4u*)(brow + g * 8);
                bq[2 * nb + 1] = *(const v4u*)(brow + 16 + g * 8);
            }
            #pragma unroll
            for (int nb = 0; nb < 4; ++nb) {
                union { v16bf v; v4u q[2]; } bfr;
                bfr.q[0] = bq[2 * nb]; bfr.q[1] = bq[2 * nb + 1];
                acc[nb] = __builtin_amdgcn_wmma_f32_16x16x32_bf16(
                    false, af.v, false, bfr.v, (short)0, acc[nb], false, false);
            }
        }
        __syncthreads();
    }

    // epilogue: C/D layout -> row = base + r + 8*g, col = nb*16 + lm
    long long baseRow = rowBase + wave * 16 + g * 8;
    #pragma unroll
    for (int nb = 0; nb < 4; ++nb) {
        int col = colBase + nb * 16 + lm;
        float bv = bias[col];
        #pragma unroll
        for (int r = 0; r < 8; ++r) {
            long long row = baseRow + r;
            if (row < (long long)M) {
                float v = acc[nb][r] + bv;
                if (relu) v = fmaxf(v, 0.0f);
                C[row * N + col] = v;
            }
        }
    }
}

// ---------------------------------------------------------------------------
__global__ __launch_bounds__(256) void cvt_bf16_kernel(
    const float* __restrict__ in, unsigned short* __restrict__ out, long long n)
{
    long long i = ((long long)blockIdx.x * 256 + threadIdx.x) * 8;
    if (i + 8 <= n) {
        float4 a = *(const float4*)(in + i);
        float4 b = *(const float4*)(in + i + 4);
        v4u o;
        o.x = (unsigned)f2bf(a.x) | ((unsigned)f2bf(a.y) << 16);
        o.y = (unsigned)f2bf(a.z) | ((unsigned)f2bf(a.w) << 16);
        o.z = (unsigned)f2bf(b.x) | ((unsigned)f2bf(b.y) << 16);
        o.w = (unsigned)f2bf(b.z) | ((unsigned)f2bf(b.w) << 16);
        *(v4u*)(out + i) = o;
    }
}

// W [K,No] f32 -> WT [No,K] bf16 (tiny; perf-irrelevant)
__global__ __launch_bounds__(256) void wt_cvt_kernel(
    const float* __restrict__ W, unsigned short* __restrict__ WT, int K, int No)
{
    int idx = blockIdx.x * 256 + threadIdx.x;
    if (idx < K * No) {
        int n = idx / K, k = idx - n * K;
        WT[idx] = f2bf(W[(long long)k * No + n]);
    }
}

__global__ void fill0_kernel(unsigned int* __restrict__ p, int n) {
    int i = blockIdx.x * blockDim.x + threadIdx.x;
    if (i < n) p[i] = 0u;
}

// segment-max: pooled >= 0 (post-relu) => float max == u32 max on bit patterns
__global__ __launch_bounds__(64) void edge_max_kernel(
    const float* __restrict__ pooled, const int* __restrict__ src,
    const int* __restrict__ dst, unsigned int* __restrict__ m, int nE, int F)
{
    int e = blockIdx.x;
    if (e >= nE) return;
    int s = src[e], d = dst[e];
    const float4* p = (const float4*)(pooled + (long long)s * F);
    unsigned int* mp = m + (long long)d * F;
    int t = threadIdx.x;                     // 64 thr x float4 = 256 feats
    float4 v = p[t];
    atomicMax(mp + t * 4 + 0, __float_as_uint(v.x));
    atomicMax(mp + t * 4 + 1, __float_as_uint(v.y));
    atomicMax(mp + t * 4 + 2, __float_as_uint(v.z));
    atomicMax(mp + t * 4 + 3, __float_as_uint(v.w));
}

// row L2-normalize f32 input, emit bf16 (next layer's GEMM operand)
__global__ __launch_bounds__(256) void l2norm_cvt_kernel(
    const float* __restrict__ in, unsigned short* __restrict__ out, int F)
{
    long long row = blockIdx.x;
    const float* hr = in + row * F;
    float ss = 0.f;
    for (int f = threadIdx.x; f < F; f += 256) { float v = hr[f]; ss += v * v; }
    for (int off = 16; off > 0; off >>= 1) ss += __shfl_down(ss, off, 32);
    __shared__ float red[9];
    if ((threadIdx.x & 31) == 0) red[threadIdx.x >> 5] = ss;
    __syncthreads();
    if (threadIdx.x == 0) {
        float s = 0.f;
        #pragma unroll
        for (int i = 0; i < 8; ++i) s += red[i];
        red[8] = 1.0f / fmaxf(sqrtf(s), 1e-12f);
    }
    __syncthreads();
    float sc = red[8];
    for (int f = threadIdx.x; f < F; f += 256)
        out[row * F + f] = f2bf(hr[f] * sc);
}

// ---------------------------------------------------------------------------
extern "C" void kernel_launch(void* const* d_in, const int* in_sizes, int n_in,
                              void* d_out, int out_size, void* d_ws, size_t ws_size,
                              hipStream_t stream) {
    (void)in_sizes; (void)n_in; (void)out_size; (void)ws_size;
    const int N = 50000, F = 256, NE = 800000;
    const long long NF = (long long)N * F;

    const float* x   = (const float*)d_in[0];
    const int*   src = (const int*)d_in[1];
    const int*   dst = (const int*)d_in[2];

    float*          pooled = (float*)d_ws;               // N*F f32 (reused as gemm2 tmp)
    float*          m      = pooled + NF;                // N*F f32
    unsigned short* hbfA   = (unsigned short*)(m + NF);  // N*F bf16
    unsigned short* hbfB   = hbfA + NF;
    unsigned short* mbf    = hbfB + NF;
    unsigned short* WpT    = mbf + NF;                   // 3x 256*256 bf16
    unsigned short* WsT    = WpT + 256 * 256;
    unsigned short* WnT    = WsT + 256 * 256;

    const int cvtBlocks = (int)((NF / 8 + 255) / 256);
    // x -> bf16
    cvt_bf16_kernel<<<cvtBlocks, 256, 0, stream>>>(x, hbfA, NF);

    unsigned short* hbf = hbfA;
    unsigned short* hnx = hbfB;
    const int dO[3] = { 256, 256, 128 };

    for (int l = 0; l < 3; ++l) {
        const float* Wp = (const float*)d_in[3 + l * 5 + 0];
        const float* bp = (const float*)d_in[3 + l * 5 + 1];
        const float* Ws = (const float*)d_in[3 + l * 5 + 2];
        const float* Wn = (const float*)d_in[3 + l * 5 + 3];
        const float* b  = (const float*)d_in[3 + l * 5 + 4];
        const int No = dO[l];

        // pre-transpose + cvt weights (tiny)
        wt_cvt_kernel<<<(F * F + 255) / 256, 256, 0, stream>>>(Wp, WpT, F, F);
        wt_cvt_kernel<<<(F * No + 255) / 256, 256, 0, stream>>>(Ws, WsT, F, No);
        wt_cvt_kernel<<<(F * No + 255) / 256, 256, 0, stream>>>(Wn, WnT, F, No);

        // pooled = relu(h @ Wp + bp)   [N,256] f32
        dim3 g1((N + TM - 1) / TM, F / TN);
        gemm_bf16_wmma_kernel<<<g1, 256, 0, stream>>>(hbf, WpT, nullptr, nullptr,
                                                      bp, pooled, N, F, F, 1);
        // m = 0 ; m[dst] = max(m[dst], pooled[src])
        fill0_kernel<<<(int)((NF + 255) / 256), 256, 0, stream>>>((unsigned int*)m, (int)NF);
        edge_max_kernel<<<NE, 64, 0, stream>>>(pooled, src, dst, (unsigned int*)m, NE, F);
        // m -> bf16
        cvt_bf16_kernel<<<cvtBlocks, 256, 0, stream>>>(m, mbf, NF);

        // out = h @ Ws + m @ Wn + b   (relu fused for layers 0,1)
        float* outF = (l < 2) ? pooled : (float*)d_out;  // pooled reusable now
        dim3 g2((N + TM - 1) / TM, No / TN);
        gemm_bf16_wmma_kernel<<<g2, 256, 0, stream>>>(hbf, WsT, mbf, WnT,
                                                      b, outF, N, F, No, l < 2 ? 1 : 0);
        if (l < 2) {
            l2norm_cvt_kernel<<<N, 256, 0, stream>>>(outF, hnx, No);
            unsigned short* tmp = hbf; hbf = hnx; hnx = tmp;
        }
    }
}
